// LSTM_12266426597813
// MI455X (gfx1250) — matrix operand order repaired and checked by reading, not compiled, not run on hardware
//
#include <hip/hip_runtime.h>
#include <hip/hip_bf16.h>

// ---------------------------------------------------------------------------
// LSTM with projection, bf16 WMMA (f32 accum) on gfx1250 (MI455X).
// INPUT=4, HIDDEN=64, PROJ=49, B=2048, T=512.
// One workgroup = 16 batch rows, 4 waves (wave32). Wave w owns hidden slice
// [16w,16w+16) of all four gates (i,f,g,o) -> c-state stays in registers.
// LDS staging buffers hold bf16 directly (A-layout-friendly): A fragments are
// two ds_load_b128 per K=32 chunk, no cvt on the critical path.
// Row stride padded to 72 bf16 (144B = 36 dwords) -> conflict-free b128 reads.
// x is prefetched with global_prefetch (8 steps ahead) and the per-step
// global_load is kept in flight across the GEMM via a sched_barrier.
// ---------------------------------------------------------------------------

typedef __attribute__((ext_vector_type(16))) __bf16 v16bf;
typedef __attribute__((ext_vector_type(8)))  __bf16 v8bf;
typedef __attribute__((ext_vector_type(4)))  __bf16 v4bf;
typedef __attribute__((ext_vector_type(8)))  float  v8f;

#define T_STEPS 512
#define B_TOT   2048
#define GN      256    // 4*HIDDEN
#define KPAD    64     // padded K (4 x-inputs + 49 proj + zeros)
#define KST     72     // LDS row stride in bf16 (144B: 16B-aligned, bank-spread)
#define XPF     8      // x prefetch distance (timesteps; 8 * 16B = 1 cacheline)

// ---------------- prep: build bf16 B-panels in workspace ----------------
// Wg  [64][256] : k 0..3 = W_ih[n][k]; k 4..52 = W_hh[n][k-4]; else 0
// Wp  [64][64]  : Wp[k][n] = (n<49) ? W_hr[n][k] : 0
// Wo  [64][64]  : Wo[k][n] = (n<49 && 4<=k<53) ? W_out[n][k-4] : 0
__global__ __launch_bounds__(256) void lstm_prep_kernel(
    const float* __restrict__ W_ih, const float* __restrict__ W_hh,
    const float* __restrict__ W_hr, const float* __restrict__ W_out,
    __bf16* __restrict__ Wg, __bf16* __restrict__ Wp, __bf16* __restrict__ Wo) {
  int idx = blockIdx.x * blockDim.x + threadIdx.x;
  if (idx < KPAD * GN) {
    int k = idx >> 8, n = idx & 255;
    float v = 0.0f;
    if (k < 4)       v = W_ih[n * 4 + k];
    else if (k < 53) v = W_hh[n * 49 + (k - 4)];
    Wg[idx] = (__bf16)v;
  }
  if (idx < KPAD * KPAD) {
    int k = idx >> 6, n = idx & 63;
    float vp = (n < 49) ? W_hr[n * 64 + k] : 0.0f;
    Wp[idx] = (__bf16)vp;
    float vo = 0.0f;
    if (n < 49 && k >= 4 && k < 53) vo = W_out[n * 49 + (k - 4)];
    Wo[idx] = (__bf16)vo;
  }
}

// ---------------- fragment loaders ----------------
// B fragment (K=32 x N=16 bf16), contiguous k order (matches A): lanes 0-15 K=0..15, lanes 16-31
// K=16..31, N = lane&15, 2 K per VGPR. (prologue-only, scalar loads are fine)
__device__ __forceinline__ v16bf load_bfrag(const __bf16* __restrict__ panel,
                                            int npad, int kbase, int n0, int lane) {
  int n = n0 + (lane & 15);
  int koff = (lane < 16) ? 0 : 16;
  v16bf f;
#pragma unroll
  for (int e = 0; e < 16; ++e)
    f[e] = panel[(kbase + koff + e) * npad + n];
  return f;
}

// A fragment (16 x K=32 bf16) straight from bf16 LDS tile [16][KST].
// lane<16 -> M=lane, K = kbase + 0..15; lane>=16 ->
// M=lane-16, K = kbase + 16..31 (matches B fragment). Both runs are contiguous bf16
// octets -> two ds_load_b128, zero conversion.
__device__ __forceinline__ v16bf load_afrag(const __bf16 (*__restrict__ buf)[KST],
                                            int kbase, int lane) {
  int m = lane & 15;
  int koff = (lane < 16) ? 0 : 16;
  const __bf16* p = &buf[m][kbase + koff];
  v8bf lo = *(const v8bf*)(p);        // K = kbase+koff+0..7
  v8bf hi = *(const v8bf*)(p + 8);    // K = kbase+koff+8..15
  return __builtin_shufflevector(lo, hi, 0, 1, 2, 3, 4, 5, 6, 7,
                                 8, 9, 10, 11, 12, 13, 14, 15);
}

__device__ __forceinline__ float fast_sigmoid(float x) {
  float e = __expf(-x);
  return __builtin_amdgcn_rcpf(1.0f + e);
}
__device__ __forceinline__ float fast_tanh(float x) {
  float e = __expf(-2.0f * x);
  return (1.0f - e) * __builtin_amdgcn_rcpf(1.0f + e);
}

// ---------------- main recurrent kernel ----------------
__global__ __launch_bounds__(128) void lstm_main_kernel(
    const float* __restrict__ x,      // [B, T, 4]
    const float* __restrict__ b_ih,   // [256]
    const float* __restrict__ b_hh,   // [256]
    const float* __restrict__ b_out,  // [49]
    const __bf16* __restrict__ Wg,    // [64][256]
    const __bf16* __restrict__ Wp,    // [64][64]
    const __bf16* __restrict__ Wo,    // [64][64]
    float* __restrict__ out) {        // [B, 49]
  __shared__ __align__(16) __bf16 hbuf[16][KST];   // cols 0..3 x_t, 4..52 h, rest 0
  __shared__ __align__(16) __bf16 hfbuf[16][KST];  // h_full (o * tanh(c))

  const int lane = threadIdx.x & 31;
  const int w    = threadIdx.x >> 5;          // wave id 0..3
  const int b0   = blockIdx.x << 4;           // batch tile base
  const int hs0  = w << 4;                    // hidden/proj slice base
  const int ncol = hs0 + (lane & 15);         // this lane's N column in slice

  // --- loop-invariant B fragments (registers) ---
  v16bf Bg[4][2];
#pragma unroll
  for (int g = 0; g < 4; ++g) {
#pragma unroll
    for (int kc = 0; kc < 2; ++kc)
      Bg[g][kc] = load_bfrag(Wg, GN, kc * 32, g * 64 + hs0, lane);
  }
  v16bf Bp0 = load_bfrag(Wp, KPAD, 0, hs0, lane);
  v16bf Bp1 = load_bfrag(Wp, KPAD, 32, hs0, lane);

  // bias broadcast per N column (same for all M rows of a D tile)
  float biasv[4];
#pragma unroll
  for (int g = 0; g < 4; ++g) {
    int n = g * 64 + hs0 + (lane & 15);
    biasv[g] = b_ih[n] + b_hh[n];
  }

  // --- init LDS: zeros (h0 = 0, pad cols stay 0 forever) ---
  for (int i = threadIdx.x; i < 16 * KST; i += 128) {
    (&hbuf[0][0])[i]  = (__bf16)0.0f;
    (&hfbuf[0][0])[i] = (__bf16)0.0f;
  }
  __syncthreads();
  // x_0 into hbuf cols 0..3
  const bool havex = (w == 0 && lane < 16);
  const float* xrow = x + (size_t)(b0 + (lane & 15)) * T_STEPS * 4;  // this lane's row
  if (havex) {
    const float4 xv = *(const float4*)(xrow);
    v4bf xb = { (__bf16)xv.x, (__bf16)xv.y, (__bf16)xv.z, (__bf16)xv.w };
    *(v4bf*)&hbuf[lane][0] = xb;
  }
  __syncthreads();

  // c state: one 16x16 f32 D-layout tile per wave (batch x hidden-slice)
  float cst[8];
#pragma unroll
  for (int e = 0; e < 8; ++e) cst[e] = 0.0f;

  for (int t = 0; t < T_STEPS; ++t) {
    // issue next-x load + far prefetch early; value is not touched until the
    // sched_barrier below, so the load stays in flight across the GEMM.
    float4 xn = make_float4(0.f, 0.f, 0.f, 0.f);
    if (havex && (t + 1) < T_STEPS) {
      xn = *(const float4*)(xrow + (size_t)(t + 1) * 4);
      if ((t + XPF) < T_STEPS)
        __builtin_prefetch(xrow + (size_t)(t + XPF) * 4, 0, 1);  // global_prefetch
    }

    // ---- gates = [x_t | h] x Wg + bias ----
    v16bf a0 = load_afrag(hbuf, 0, lane);
    v16bf a1 = load_afrag(hbuf, 32, lane);
    v8f gd[4];
#pragma unroll
    for (int g = 0; g < 4; ++g) {
      v8f acc;
#pragma unroll
      for (int e = 0; e < 8; ++e) acc[e] = biasv[g];
      acc = __builtin_amdgcn_wmma_f32_16x16x32_bf16(false, a0, false, Bg[g][0],
                                                    (short)0, acc, false, false);
      acc = __builtin_amdgcn_wmma_f32_16x16x32_bf16(false, a1, false, Bg[g][1],
                                                    (short)0, acc, false, false);
      gd[g] = acc;
    }
    // ---- activations + c update + h_full, register-resident ----
#pragma unroll
    for (int e = 0; e < 8; ++e) {
      int M = (lane < 16) ? e : e + 8;
      float iv = fast_sigmoid(gd[0][e]);
      float fv = fast_sigmoid(gd[1][e]);
      float gv = fast_tanh(gd[2][e]);
      float ov = fast_sigmoid(gd[3][e]);
      cst[e] = fv * cst[e] + iv * gv;
      hfbuf[M][ncol] = (__bf16)(ov * fast_tanh(cst[e]));  // ds_store_b16
    }
    __syncthreads();   // hfbuf ready; also: all hbuf reads of this step done

    // ---- projection: h = h_full x W_hr^T (this wave: proj cols hs0..hs0+15)
    v16bf pa0 = load_afrag(hfbuf, 0, lane);
    v16bf pa1 = load_afrag(hfbuf, 32, lane);
    v8f pd;
#pragma unroll
    for (int e = 0; e < 8; ++e) pd[e] = 0.0f;
    pd = __builtin_amdgcn_wmma_f32_16x16x32_bf16(false, pa0, false, Bp0,
                                                 (short)0, pd, false, false);
    pd = __builtin_amdgcn_wmma_f32_16x16x32_bf16(false, pa1, false, Bp1,
                                                 (short)0, pd, false, false);
    // scatter new h (as bf16) into hbuf cols 4..52 (cols 53..63 stay zero)
    if (ncol < 49) {
#pragma unroll
      for (int e = 0; e < 8; ++e) {
        int M = (lane < 16) ? e : e + 8;
        hbuf[M][4 + ncol] = (__bf16)pd[e];
      }
    }
    // keep the x-load wait (and its cvt) below this point: the load issued at
    // the top of the iteration has the whole GEMM body to complete.
#if __has_builtin(__builtin_amdgcn_sched_barrier)
    __builtin_amdgcn_sched_barrier(0);
#endif
    // commit prefetched x_{t+1} into hbuf cols 0..3
    if (havex && (t + 1) < T_STEPS) {
      v4bf xb = { (__bf16)xn.x, (__bf16)xn.y, (__bf16)xn.z, (__bf16)xn.w };
      *(v4bf*)&hbuf[lane][0] = xb;
    }
    __syncthreads();   // hbuf ready for next step
  }

  // ---- final: out = h_T x W_out^T + b_out (Wo has x-rows zeroed, reuse hbuf)
  v16bf oa0 = load_afrag(hbuf, 0, lane);
  v16bf oa1 = load_afrag(hbuf, 32, lane);
  v16bf Bo0 = load_bfrag(Wo, KPAD, 0, hs0, lane);
  v16bf Bo1 = load_bfrag(Wo, KPAD, 32, hs0, lane);
  float bo = (ncol < 49) ? b_out[ncol] : 0.0f;
  v8f od;
#pragma unroll
  for (int e = 0; e < 8; ++e) od[e] = bo;
  od = __builtin_amdgcn_wmma_f32_16x16x32_bf16(false, oa0, false, Bo0,
                                               (short)0, od, false, false);
  od = __builtin_amdgcn_wmma_f32_16x16x32_bf16(false, oa1, false, Bo1,
                                               (short)0, od, false, false);
  if (ncol < 49) {
#pragma unroll
    for (int e = 0; e < 8; ++e) {
      int M = (lane < 16) ? e : e + 8;
      out[(size_t)(b0 + M) * 49 + ncol] = od[e];
    }
  }
}

// ---------------------------------------------------------------------------
extern "C" void kernel_launch(void* const* d_in, const int* in_sizes, int n_in,
                              void* d_out, int out_size, void* d_ws, size_t ws_size,
                              hipStream_t stream) {
  const float* x     = (const float*)d_in[0];
  const float* W_ih  = (const float*)d_in[1];
  const float* W_hh  = (const float*)d_in[2];
  const float* b_ih  = (const float*)d_in[3];
  const float* b_hh  = (const float*)d_in[4];
  const float* W_hr  = (const float*)d_in[5];
  const float* W_out = (const float*)d_in[6];
  const float* b_out = (const float*)d_in[7];
  float* out = (float*)d_out;

  // workspace layout (bf16 panels): Wg [64][256], Wp [64][64], Wo [64][64]
  char* ws = (char*)d_ws;
  __bf16* Wg = (__bf16*)(ws);
  __bf16* Wp = (__bf16*)(ws + (size_t)KPAD * GN * 2);
  __bf16* Wo = (__bf16*)(ws + (size_t)KPAD * GN * 2 + (size_t)KPAD * KPAD * 2);

  lstm_prep_kernel<<<dim3((KPAD * GN + 255) / 256), dim3(256), 0, stream>>>(
      W_ih, W_hh, W_hr, W_out, Wg, Wp, Wo);

  lstm_main_kernel<<<dim3(B_TOT / 16), dim3(128), 0, stream>>>(
      x, b_ih, b_hh, b_out, Wg, Wp, Wo, out);
}
